// Graph_25211458027671
// MI455X (gfx1250) — matrix operand order repaired and checked by reading, not compiled, not run on hardware
//
#include <hip/hip_runtime.h>
#include <hip/hip_bf16.h>
#include <math.h>

// ---------------------------------------------------------------------------
// ProteinMPNN-style encoder layer for gfx1250 (MI455X).
// bf16 WMMA (v_wmma_f32_16x16x32_bf16) for all message-MLP GEMMs,
// VALU for the tiny per-node FFN + LayerNorms. wave32 throughout.
// ---------------------------------------------------------------------------

typedef __attribute__((ext_vector_type(16))) __bf16 v16bf;
typedef __attribute__((ext_vector_type(8)))  float  v8f;
typedef __attribute__((ext_vector_type(4)))  unsigned int u32x4;

#define B_    2
#define N_    1024
#define K_    48
#define H_    128
#define CAT_  384
#define FF_   512
#define SCALE_ 48.0f
#define EPS_  1e-5f

__device__ __forceinline__ float gelu_exact(float x) {
  return 0.5f * x * (1.0f + erff(x * 0.70710678118654752440f));
}

union FragU { u32x4 q[2]; v16bf v; };

// A-matrix 16x32 bf16 fragment from LDS (row-major, stride in elements).
// Layout (ISA 7.12.2): lanes 0-15 row M=lane, K blocks [0..7] then [16..23];
// lanes 16-31 K blocks [8..15] then [24..31].
__device__ __forceinline__ v16bf load_a_frag_lds(const __bf16* base, int stride,
                                                 int m0, int k0, int lane) {
  int row = m0 + (lane & 15);
  int kb  = k0 + ((lane >> 4) << 3);
  FragU f;
  f.q[0] = *reinterpret_cast<const u32x4*>(base + row * stride + kb);
  f.q[1] = *reinterpret_cast<const u32x4*>(base + row * stride + kb + 16);
  return f.v;
}

// B-matrix 32x16 bf16 fragment from global weights stored [N][K] (row-major).
// Layout: lanes 0-15 col N=lane hold K blocks [0..7] then [16..23]; lanes 16-31 [8..15] then [24..31].
__device__ __forceinline__ v16bf load_b_frag_glb(const __bf16* Wt, int stride,
                                                 int n0, int k0, int lane) {
  int row = n0 + (lane & 15);
  int kb  = k0 + ((lane >> 4) << 3);
  FragU f;
  f.q[0] = *reinterpret_cast<const u32x4*>(Wt + row * stride + kb);
  f.q[1] = *reinterpret_cast<const u32x4*>(Wt + row * stride + kb + 16);
  return f.v;
}

// C/D 16x16 f32: VGPR r -> (M = m0 + r + 8*(lane>=16), N = n0 + lane%16)
__device__ __forceinline__ void store_tile_gelu(__bf16* dst, int stride,
                                                int m0, int n0, int lane,
                                                v8f acc, float bias) {
  int col = n0 + (lane & 15);
  int hi  = (lane >> 4) << 3;
#pragma unroll
  for (int r = 0; r < 8; ++r)
    dst[(m0 + r + hi) * stride + col] = (__bf16)gelu_exact(acc[r] + bias);
}

__device__ __forceinline__ float reduce128(float* red, int tid, float v) {
  if (tid < 128) red[tid] = v;
  __syncthreads();
#pragma unroll
  for (int s = 64; s > 0; s >>= 1) {
    if (tid < s) red[tid] += red[tid + s];
    __syncthreads();
  }
  float r = red[0];
  __syncthreads();
  return r;
}

// ---------------------------------------------------------------------------
// Weight transpose+convert: src [din][dout] f32 -> dst [dout][din] bf16
// ---------------------------------------------------------------------------
__global__ void transpose_bf16_kernel(const float* __restrict__ src,
                                      __bf16* __restrict__ dst,
                                      int din, int dout) {
  int i = blockIdx.x * blockDim.x + threadIdx.x;
  if (i < din * dout) {
    int o  = i / din;
    int in = i - o * din;
    dst[o * din + in] = (__bf16)src[in * dout + o];
  }
}

// ---------------------------------------------------------------------------
// Node path: message MLP + masked sum + LN1 + FFN + LN2 + mask_V
// One block per (b,n). 256 threads = 8 wave32.
// ---------------------------------------------------------------------------
__global__ __launch_bounds__(256)
void node_kernel(const float* __restrict__ hV, const float* __restrict__ hE,
                 const int* __restrict__ Eidx,
                 const float* __restrict__ maskV, const float* __restrict__ maskAtt,
                 const __bf16* __restrict__ W1t, const float* __restrict__ W1b,
                 const __bf16* __restrict__ W2t, const float* __restrict__ W2b,
                 const __bf16* __restrict__ W3t, const float* __restrict__ W3b,
                 const float* __restrict__ Winw, const float* __restrict__ Winb,
                 const float* __restrict__ Woutw, const float* __restrict__ Woutb,
                 const float* __restrict__ g1, const float* __restrict__ b1,
                 const float* __restrict__ g2, const float* __restrict__ b2,
                 float* __restrict__ hVn_out, __bf16* __restrict__ hVn_bf) {
  __shared__ __align__(16) __bf16 A0[K_ * CAT_];    // 36864 B (hEV, bf16)
  __shared__ __align__(16) __bf16 Cbuf[K_ * H_];    // 12288 B (ping buffer)
  __shared__ float dhv[H_];
  __shared__ float red[H_];
  __shared__ float h1s[H_];
  __shared__ float hid[FF_];
  __shared__ float pp[256];
  __shared__ float pm[K_];
  __shared__ int   nbr[K_];

  const int bn   = blockIdx.x;
  const int tid  = threadIdx.x;
  const int lane = tid & 31;
  const int w    = tid >> 5;
  const int n0   = w * 16;
  const long rowV = (long)bn * H_;
  const long rowE = (long)bn * (long)K_ * H_;
  const long bbase = (long)(bn / N_) * N_;

  if (tid < K_) {
    nbr[tid] = Eidx[(long)bn * K_ + tid];
    pm[tid]  = maskAtt[(long)bn * K_ + tid];
  }
  __syncthreads();

  // build hEV = [hV_i | hE_ij | hV_j]  -> bf16 in LDS
  for (int e = tid; e < K_ * CAT_; e += 256) {
    int j = e / CAT_, c = e - j * CAT_;
    float v;
    if (c < H_)           v = hV[rowV + c];
    else if (c < 2 * H_)  v = hE[rowE + (long)j * H_ + (c - H_)];
    else                  v = hV[(bbase + nbr[j]) * H_ + (c - 2 * H_)];
    A0[e] = (__bf16)v;
  }
  __syncthreads();

  // ---- GEMM1: [48x384] @ [384x128] ----
  v8f acc0 = {}, acc1 = {}, acc2 = {};
  for (int k0 = 0; k0 < CAT_; k0 += 32) {
    v16bf bf = load_b_frag_glb(W1t, CAT_, n0, k0, lane);
    v16bf a0 = load_a_frag_lds(A0, CAT_, 0,  k0, lane);
    v16bf a1 = load_a_frag_lds(A0, CAT_, 16, k0, lane);
    v16bf a2 = load_a_frag_lds(A0, CAT_, 32, k0, lane);
    acc0 = __builtin_amdgcn_wmma_f32_16x16x32_bf16(false, a0, false, bf, (short)0, acc0, false, false);
    acc1 = __builtin_amdgcn_wmma_f32_16x16x32_bf16(false, a1, false, bf, (short)0, acc1, false, false);
    acc2 = __builtin_amdgcn_wmma_f32_16x16x32_bf16(false, a2, false, bf, (short)0, acc2, false, false);
  }
  {
    float bias = W1b[n0 + (lane & 15)];
    store_tile_gelu(Cbuf, H_, 0,  n0, lane, acc0, bias);
    store_tile_gelu(Cbuf, H_, 16, n0, lane, acc1, bias);
    store_tile_gelu(Cbuf, H_, 32, n0, lane, acc2, bias);
  }
  __syncthreads();

  // ---- GEMM2: [48x128] @ [128x128] ----
  acc0 = (v8f){}; acc1 = (v8f){}; acc2 = (v8f){};
  for (int k0 = 0; k0 < H_; k0 += 32) {
    v16bf bf = load_b_frag_glb(W2t, H_, n0, k0, lane);
    v16bf a0 = load_a_frag_lds(Cbuf, H_, 0,  k0, lane);
    v16bf a1 = load_a_frag_lds(Cbuf, H_, 16, k0, lane);
    v16bf a2 = load_a_frag_lds(Cbuf, H_, 32, k0, lane);
    acc0 = __builtin_amdgcn_wmma_f32_16x16x32_bf16(false, a0, false, bf, (short)0, acc0, false, false);
    acc1 = __builtin_amdgcn_wmma_f32_16x16x32_bf16(false, a1, false, bf, (short)0, acc1, false, false);
    acc2 = __builtin_amdgcn_wmma_f32_16x16x32_bf16(false, a2, false, bf, (short)0, acc2, false, false);
  }
  __syncthreads();   // all reads of Cbuf done before rewrite
  {
    float bias = W2b[n0 + (lane & 15)];
    store_tile_gelu(Cbuf, H_, 0,  n0, lane, acc0, bias);
    store_tile_gelu(Cbuf, H_, 16, n0, lane, acc1, bias);
    store_tile_gelu(Cbuf, H_, 32, n0, lane, acc2, bias);
  }
  __syncthreads();

  // ---- GEMM3 + masked column sum ----
  acc0 = (v8f){}; acc1 = (v8f){}; acc2 = (v8f){};
  for (int k0 = 0; k0 < H_; k0 += 32) {
    v16bf bf = load_b_frag_glb(W3t, H_, n0, k0, lane);
    v16bf a0 = load_a_frag_lds(Cbuf, H_, 0,  k0, lane);
    v16bf a1 = load_a_frag_lds(Cbuf, H_, 16, k0, lane);
    v16bf a2 = load_a_frag_lds(Cbuf, H_, 32, k0, lane);
    acc0 = __builtin_amdgcn_wmma_f32_16x16x32_bf16(false, a0, false, bf, (short)0, acc0, false, false);
    acc1 = __builtin_amdgcn_wmma_f32_16x16x32_bf16(false, a1, false, bf, (short)0, acc1, false, false);
    acc2 = __builtin_amdgcn_wmma_f32_16x16x32_bf16(false, a2, false, bf, (short)0, acc2, false, false);
  }
  {
    int col   = n0 + (lane & 15);
    int hi    = (lane >> 4) << 3;
    float bias = W3b[col];
    float partial = 0.f;
#pragma unroll
    for (int r = 0; r < 8; ++r) {
      partial += (acc0[r] + bias) * pm[r + hi];
      partial += (acc1[r] + bias) * pm[16 + r + hi];
      partial += (acc2[r] + bias) * pm[32 + r + hi];
    }
    partial += __shfl_xor(partial, 16, 32);   // combine lane halves (same col)
    if (lane < 16) dhv[col] = partial;        // each wave owns 16 distinct cols
  }
  __syncthreads();

  // ---- LN1 ----
  float xval = 0.f;
  if (tid < H_) xval = hV[rowV + tid] + dhv[tid] * (1.0f / SCALE_);
  float mean = reduce128(red, tid, xval) * (1.0f / H_);
  float d = xval - mean;
  float var = reduce128(red, tid, d * d) * (1.0f / H_);
  float inv = rsqrtf(var + EPS_);
  if (tid < H_) h1s[tid] = d * inv * g1[tid] + b1[tid];
  __syncthreads();

  // ---- FFN (GEMV, VALU) ----
  for (int u = tid; u < FF_; u += 256) {
    float s = Winb[u];
    for (int j = 0; j < H_; ++j) s += h1s[j] * Winw[j * FF_ + u];
    hid[u] = gelu_exact(s);
  }
  __syncthreads();
  {
    int u = tid & 127, half = tid >> 7;
    float s = 0.f;
    int j0 = half * 256;
    for (int j = j0; j < j0 + 256; ++j) s += hid[j] * Woutw[j * H_ + u];
    pp[tid] = s;
  }
  __syncthreads();

  // ---- LN2 + mask_V ----
  float yv = 0.f;
  if (tid < H_) yv = h1s[tid] + pp[tid] + pp[tid + 128] + Woutb[tid];
  float mean2 = reduce128(red, tid, yv) * (1.0f / H_);
  float d2 = yv - mean2;
  float var2 = reduce128(red, tid, d2 * d2) * (1.0f / H_);
  float inv2 = rsqrtf(var2 + EPS_);
  if (tid < H_) {
    float o = (d2 * inv2 * g2[tid] + b2[tid]) * maskV[bn];
    hVn_out[rowV + tid] = o;
    hVn_bf[rowV + tid]  = (__bf16)o;
  }
}

// ---------------------------------------------------------------------------
// Edge path: message MLP on updated nodes + LN3.  One block per (b,n).
// ---------------------------------------------------------------------------
__global__ __launch_bounds__(256)
void edge_kernel(const float* __restrict__ hE, const int* __restrict__ Eidx,
                 const __bf16* __restrict__ hVnb,
                 const __bf16* __restrict__ W11t, const float* __restrict__ W11b,
                 const __bf16* __restrict__ W12t, const float* __restrict__ W12b,
                 const __bf16* __restrict__ W13t, const float* __restrict__ W13b,
                 const float* __restrict__ g3, const float* __restrict__ b3,
                 float* __restrict__ hEn_out) {
  __shared__ __align__(16) char smemA[K_ * CAT_ * 2];   // 36864 B
  __bf16* A0 = (__bf16*)smemA;                          // phase 1 (hEV bf16)
  float*  C3 = (float*)smemA;                           // phase 2 (24576 B)
  __shared__ __align__(16) __bf16 Cbuf[K_ * H_];
  __shared__ int nbr[K_];

  const int bn   = blockIdx.x;
  const int tid  = threadIdx.x;
  const int lane = tid & 31;
  const int w    = tid >> 5;
  const int n0   = w * 16;
  const long rowV = (long)bn * H_;
  const long rowE = (long)bn * (long)K_ * H_;
  const long bbase = (long)(bn / N_) * N_;

  if (tid < K_) nbr[tid] = Eidx[(long)bn * K_ + tid];
  __syncthreads();

  for (int e = tid; e < K_ * CAT_; e += 256) {
    int j = e / CAT_, c = e - j * CAT_;
    __bf16 v;
    if (c < H_)           v = hVnb[rowV + c];
    else if (c < 2 * H_)  v = (__bf16)hE[rowE + (long)j * H_ + (c - H_)];
    else                  v = hVnb[(bbase + nbr[j]) * H_ + (c - 2 * H_)];
    A0[e] = v;
  }
  __syncthreads();

  // ---- GEMM1 (W11) ----
  v8f acc0 = {}, acc1 = {}, acc2 = {};
  for (int k0 = 0; k0 < CAT_; k0 += 32) {
    v16bf bf = load_b_frag_glb(W11t, CAT_, n0, k0, lane);
    v16bf a0 = load_a_frag_lds(A0, CAT_, 0,  k0, lane);
    v16bf a1 = load_a_frag_lds(A0, CAT_, 16, k0, lane);
    v16bf a2 = load_a_frag_lds(A0, CAT_, 32, k0, lane);
    acc0 = __builtin_amdgcn_wmma_f32_16x16x32_bf16(false, a0, false, bf, (short)0, acc0, false, false);
    acc1 = __builtin_amdgcn_wmma_f32_16x16x32_bf16(false, a1, false, bf, (short)0, acc1, false, false);
    acc2 = __builtin_amdgcn_wmma_f32_16x16x32_bf16(false, a2, false, bf, (short)0, acc2, false, false);
  }
  {
    float bias = W11b[n0 + (lane & 15)];
    store_tile_gelu(Cbuf, H_, 0,  n0, lane, acc0, bias);
    store_tile_gelu(Cbuf, H_, 16, n0, lane, acc1, bias);
    store_tile_gelu(Cbuf, H_, 32, n0, lane, acc2, bias);
  }
  __syncthreads();

  // ---- GEMM2 (W12) ----
  acc0 = (v8f){}; acc1 = (v8f){}; acc2 = (v8f){};
  for (int k0 = 0; k0 < H_; k0 += 32) {
    v16bf bf = load_b_frag_glb(W12t, H_, n0, k0, lane);
    v16bf a0 = load_a_frag_lds(Cbuf, H_, 0,  k0, lane);
    v16bf a1 = load_a_frag_lds(Cbuf, H_, 16, k0, lane);
    v16bf a2 = load_a_frag_lds(Cbuf, H_, 32, k0, lane);
    acc0 = __builtin_amdgcn_wmma_f32_16x16x32_bf16(false, a0, false, bf, (short)0, acc0, false, false);
    acc1 = __builtin_amdgcn_wmma_f32_16x16x32_bf16(false, a1, false, bf, (short)0, acc1, false, false);
    acc2 = __builtin_amdgcn_wmma_f32_16x16x32_bf16(false, a2, false, bf, (short)0, acc2, false, false);
  }
  __syncthreads();
  {
    float bias = W12b[n0 + (lane & 15)];
    store_tile_gelu(Cbuf, H_, 0,  n0, lane, acc0, bias);
    store_tile_gelu(Cbuf, H_, 16, n0, lane, acc1, bias);
    store_tile_gelu(Cbuf, H_, 32, n0, lane, acc2, bias);
  }
  __syncthreads();

  // ---- GEMM3 (W13) -> C3 f32 (aliases A0; all A0 reads are long done) ----
  acc0 = (v8f){}; acc1 = (v8f){}; acc2 = (v8f){};
  for (int k0 = 0; k0 < H_; k0 += 32) {
    v16bf bf = load_b_frag_glb(W13t, H_, n0, k0, lane);
    v16bf a0 = load_a_frag_lds(Cbuf, H_, 0,  k0, lane);
    v16bf a1 = load_a_frag_lds(Cbuf, H_, 16, k0, lane);
    v16bf a2 = load_a_frag_lds(Cbuf, H_, 32, k0, lane);
    acc0 = __builtin_amdgcn_wmma_f32_16x16x32_bf16(false, a0, false, bf, (short)0, acc0, false, false);
    acc1 = __builtin_amdgcn_wmma_f32_16x16x32_bf16(false, a1, false, bf, (short)0, acc1, false, false);
    acc2 = __builtin_amdgcn_wmma_f32_16x16x32_bf16(false, a2, false, bf, (short)0, acc2, false, false);
  }
  {
    int col = n0 + (lane & 15);
    int hi  = (lane >> 4) << 3;
    float bias = W13b[col];
#pragma unroll
    for (int r = 0; r < 8; ++r) {
      C3[(r + hi) * H_ + col]      = acc0[r] + bias;
      C3[(16 + r + hi) * H_ + col] = acc1[r] + bias;
      C3[(32 + r + hi) * H_ + col] = acc2[r] + bias;
    }
  }
  __syncthreads();

  // ---- LN3 per edge row (wave32 shuffle reduction, 6 rows per wave) ----
  for (int r = w; r < K_; r += 8) {
    float x[4];
    float s = 0.f;
#pragma unroll
    for (int t = 0; t < 4; ++t) {
      int c = lane + 32 * t;
      x[t] = C3[r * H_ + c] + hE[rowE + (long)r * H_ + c];
      s += x[t];
    }
#pragma unroll
    for (int o = 16; o > 0; o >>= 1) s += __shfl_xor(s, o, 32);
    float mean = s * (1.0f / H_);
    float dsq = 0.f;
#pragma unroll
    for (int t = 0; t < 4; ++t) { float d = x[t] - mean; dsq += d * d; }
#pragma unroll
    for (int o = 16; o > 0; o >>= 1) dsq += __shfl_xor(dsq, o, 32);
    float inv = rsqrtf(dsq * (1.0f / H_) + EPS_);
#pragma unroll
    for (int t = 0; t < 4; ++t) {
      int c = lane + 32 * t;
      hEn_out[rowE + (long)r * H_ + c] = (x[t] - mean) * inv * g3[c] + b3[c];
    }
  }
}

// ---------------------------------------------------------------------------
extern "C" void kernel_launch(void* const* d_in, const int* in_sizes, int n_in,
                              void* d_out, int out_size, void* d_ws, size_t ws_size,
                              hipStream_t stream) {
  const float* hV      = (const float*)d_in[0];
  const float* hE      = (const float*)d_in[1];
  const int*   Eidx    = (const int*)d_in[2];
  const float* maskV   = (const float*)d_in[3];
  const float* maskAtt = (const float*)d_in[4];
  const float* W1w  = (const float*)d_in[5];  const float* W1b  = (const float*)d_in[6];
  const float* W2w  = (const float*)d_in[7];  const float* W2b  = (const float*)d_in[8];
  const float* W3w  = (const float*)d_in[9];  const float* W3b  = (const float*)d_in[10];
  const float* W11w = (const float*)d_in[11]; const float* W11b = (const float*)d_in[12];
  const float* W12w = (const float*)d_in[13]; const float* W12b = (const float*)d_in[14];
  const float* W13w = (const float*)d_in[15]; const float* W13b = (const float*)d_in[16];
  const float* Winw = (const float*)d_in[17]; const float* Winb = (const float*)d_in[18];
  const float* Woutw = (const float*)d_in[19]; const float* Woutb = (const float*)d_in[20];
  const float* g1 = (const float*)d_in[21]; const float* b1 = (const float*)d_in[22];
  const float* g2 = (const float*)d_in[23]; const float* b2 = (const float*)d_in[24];
  const float* g3 = (const float*)d_in[25]; const float* b3 = (const float*)d_in[26];

  // workspace layout (bf16 transposed weights + bf16 copy of updated nodes)
  char* ws = (char*)d_ws;
  __bf16* W1t  = (__bf16*)ws;  ws += (size_t)H_ * CAT_ * 2;
  __bf16* W2t  = (__bf16*)ws;  ws += (size_t)H_ * H_ * 2;
  __bf16* W3t  = (__bf16*)ws;  ws += (size_t)H_ * H_ * 2;
  __bf16* W11t = (__bf16*)ws;  ws += (size_t)H_ * CAT_ * 2;
  __bf16* W12t = (__bf16*)ws;  ws += (size_t)H_ * H_ * 2;
  __bf16* W13t = (__bf16*)ws;  ws += (size_t)H_ * H_ * 2;
  __bf16* hVnb = (__bf16*)ws;  ws += (size_t)B_ * N_ * H_ * 2;

  float* hVn_out = (float*)d_out;                       // [B,N,H]
  float* hEn_out = (float*)d_out + (size_t)B_ * N_ * H_; // [B,N,K,H]

  // convert/transpose weights to bf16 [out][in]
  transpose_bf16_kernel<<<(CAT_ * H_ + 255) / 256, 256, 0, stream>>>(W1w,  W1t,  CAT_, H_);
  transpose_bf16_kernel<<<(H_  * H_ + 255) / 256, 256, 0, stream>>>(W2w,  W2t,  H_,   H_);
  transpose_bf16_kernel<<<(H_  * H_ + 255) / 256, 256, 0, stream>>>(W3w,  W3t,  H_,   H_);
  transpose_bf16_kernel<<<(CAT_ * H_ + 255) / 256, 256, 0, stream>>>(W11w, W11t, CAT_, H_);
  transpose_bf16_kernel<<<(H_  * H_ + 255) / 256, 256, 0, stream>>>(W12w, W12t, H_,   H_);
  transpose_bf16_kernel<<<(H_  * H_ + 255) / 256, 256, 0, stream>>>(W13w, W13t, H_,   H_);

  node_kernel<<<B_ * N_, 256, 0, stream>>>(
      hV, hE, Eidx, maskV, maskAtt,
      W1t, W1b, W2t, W2b, W3t, W3b,
      Winw, Winb, Woutw, Woutb,
      g1, b1, g2, b2, hVn_out, hVnb);

  edge_kernel<<<B_ * N_, 256, 0, stream>>>(
      hE, Eidx, hVnb,
      W11t, W11b, W12t, W12b, W13t, W13b,
      g3, b3, hEn_out);
}
